// Global_Reactivity_Attention_39883066311263
// MI455X (gfx1250) — matrix operand order repaired and checked by reading, not compiled, not run on hardware
//
#include <hip/hip_runtime.h>
#include <math.h>

#define NTOK   3072
#define DMODEL 512
#define NH     8
#define NPK    8
#define DKH    64
#define NL     3

#define BM 64
#define BN 64
#define BK 32

typedef __attribute__((ext_vector_type(16))) _Float16 v16h;
typedef __attribute__((ext_vector_type(8)))  float    v8f;

__device__ __forceinline__ float sigmoid_f(float v) {
    return 1.0f / (1.0f + __expf(-v));
}
__device__ __forceinline__ float gelu_f(float v) {
    const float c = 0.7978845608028654f;
    return 0.5f * v * (1.0f + tanhf(c * (v + 0.044715f * v * v * v)));
}

// ---------------------------------------------------------------------------
// LayerNorm over D=512, one block (256 threads) per row.
// ---------------------------------------------------------------------------
__global__ __launch_bounds__(256) void layernorm_kernel(
    const float* __restrict__ x, const float* __restrict__ s,
    const float* __restrict__ b, float* __restrict__ out) {
    const int row = blockIdx.x;
    const int t = threadIdx.x;
    const float* xr = x + (long)row * DMODEL;
    __shared__ float red[256];

    float v0 = xr[t];
    float v1 = xr[t + 256];
    red[t] = v0 + v1;
    __syncthreads();
    for (int off = 128; off > 0; off >>= 1) {
        if (t < off) red[t] += red[t + off];
        __syncthreads();
    }
    float mean = red[0] * (1.0f / DMODEL);
    __syncthreads();
    float d0 = v0 - mean, d1 = v1 - mean;
    red[t] = d0 * d0 + d1 * d1;
    __syncthreads();
    for (int off = 128; off > 0; off >>= 1) {
        if (t < off) red[t] += red[t + off];
        __syncthreads();
    }
    float inv = rsqrtf(red[0] * (1.0f / DMODEL) + 1e-6f);
    float* orow = out + (long)row * DMODEL;
    orow[t]       = d0 * inv * s[t]       + b[t];
    orow[t + 256] = d1 * inv * s[t + 256] + b[t + 256];
}

// ---------------------------------------------------------------------------
// rq[h][i][p] = sum_d q[i][h*64+d] * relk[p][d]   (block per i, 64 threads)
// ---------------------------------------------------------------------------
__global__ __launch_bounds__(64) void relproj_kernel(
    const float* __restrict__ q, const float* __restrict__ relk,
    float* __restrict__ rq) {
    const int i = blockIdx.x;
    const int t = threadIdx.x;
    const int h = t >> 3, p = t & 7;
    const float* qp = q + (long)i * DMODEL + h * DKH;
    const float* rp = relk + p * DKH;
    float acc = 0.0f;
#pragma unroll 8
    for (int d = 0; d < DKH; ++d) acc += qp[d] * rp[d];
    rq[((long)h * NTOK + i) * NPK + p] = acc;
}

// ---------------------------------------------------------------------------
// Generic WMMA GEMM (NN): 64x64 block tile, 8 wave32s, BK=32.
// ACT: 0 none, 1 sigmoid(+bias), 2 gelu(+bias), 3 elementwise-multiply by
//      `resid` pointer (gate fusion).  resid (add or mul) is z-offset by zC.
// ---------------------------------------------------------------------------
template <int ACT>
__global__ __launch_bounds__(256) void gemm_wmma(
    const float* __restrict__ A, const float* __restrict__ B,
    const float* __restrict__ bias, const float* __restrict__ resid,
    float* __restrict__ C, int K, int lda, int ldb, int ldc,
    long zA, long zB, long zC) {
    const int z = blockIdx.z;
    A += (long)z * zA;
    B += (long)z * zB;
    C += (long)z * zC;
    const float* residz = resid ? resid + (long)z * zC : nullptr;
    const int row0 = blockIdx.y * BM;
    const int col0 = blockIdx.x * BN;

    __shared__ alignas(16) _Float16 sA[BM * BK];      // [m][k]
    __shared__ alignas(16) _Float16 sB[BN * BK];      // [n][k]  (K-transposed)

    const int t = threadIdx.x;
    const int lane = t & 31;
    const int wid = t >> 5;
    const int wm = wid & 3;   // 0..3 -> 16-row stripe
    const int wn = wid >> 2;  // 0..1 -> 32-col stripe

    v8f acc0 = {0.f, 0.f, 0.f, 0.f, 0.f, 0.f, 0.f, 0.f};
    v8f acc1 = acc0;

    const int nk = K / BK;
    for (int kt = 0; kt < nk; ++kt) {
        const int k0 = kt * BK;
        // ---- stage A tile (64x32 f32 -> f16) : 2 float4 per thread
#pragma unroll
        for (int i = 0; i < 2; ++i) {
            int f = t + i * 256;
            int r = f >> 3, c = (f & 7) << 2;
            float4 d = *(const float4*)(A + (long)(row0 + r) * lda + k0 + c);
            _Float16* dst = sA + r * BK + c;
            dst[0] = (_Float16)d.x; dst[1] = (_Float16)d.y;
            dst[2] = (_Float16)d.z; dst[3] = (_Float16)d.w;
        }
        // ---- stage B tile (32x64 f32 row-major -> sB[n][k] f16)
#pragma unroll
        for (int i = 0; i < 2; ++i) {
            int f = t + i * 256;
            int k = f >> 4, n = (f & 15) << 2;
            float4 d = *(const float4*)(B + (long)(k0 + k) * ldb + col0 + n);
            sB[(n + 0) * BK + k] = (_Float16)d.x;
            sB[(n + 1) * BK + k] = (_Float16)d.y;
            sB[(n + 2) * BK + k] = (_Float16)d.z;
            sB[(n + 3) * BK + k] = (_Float16)d.w;
        }
        if (kt + 1 < nk) {  // prefetch next k-tile (global_prefetch_b8)
            __builtin_prefetch(A + (long)(row0 + (t & 63)) * lda + k0 + BK, 0, 3);
            __builtin_prefetch(B + (long)(k0 + BK + (t & 31)) * ldb + col0, 0, 3);
        }
        __syncthreads();

        // ---- fragments per ISA 7.12.2 layouts
        union { v16h v; uint4 u[2]; } fa, fb0, fb1;
        {
            const int r = wm * 16 + (lane & 15);
            const int kb = (lane < 16) ? 0 : 8;
            const _Float16* p = sA + r * BK;
            fa.u[0] = *(const uint4*)(p + kb);
            fa.u[1] = *(const uint4*)(p + 16 + kb);
        }
        {
            const int kb = (lane < 16) ? 0 : 8;
            const _Float16* p0 = sB + (wn * 32 + (lane & 15)) * BK + kb;
            fb0.u[0] = *(const uint4*)(p0);
            fb0.u[1] = *(const uint4*)(p0 + 16);
            const _Float16* p1 = sB + (wn * 32 + 16 + (lane & 15)) * BK + kb;
            fb1.u[0] = *(const uint4*)(p1);
            fb1.u[1] = *(const uint4*)(p1 + 16);
        }
        acc0 = __builtin_amdgcn_wmma_f32_16x16x32_f16(false, fa.v, false, fb0.v,
                                                      (short)0, acc0, false, false);
        acc1 = __builtin_amdgcn_wmma_f32_16x16x32_f16(false, fa.v, false, fb1.v,
                                                      (short)0, acc1, false, false);
        __syncthreads();
    }

    // ---- epilogue (C/D layout: VGPR r -> row r (+8 for upper lane half))
    const int rbase = row0 + wm * 16 + ((lane < 16) ? 0 : 8);
    const int c0 = col0 + wn * 32 + (lane & 15);
    const int c1 = c0 + 16;
#pragma unroll
    for (int r = 0; r < 8; ++r) {
        const int row = rbase + r;
        float v0 = acc0[r], v1 = acc1[r];
        if (bias) { v0 += bias[c0]; v1 += bias[c1]; }
        if (ACT == 1) { v0 = sigmoid_f(v0); v1 = sigmoid_f(v1); }
        else if (ACT == 2) { v0 = gelu_f(v0); v1 = gelu_f(v1); }
        if (ACT == 3) {  // fused gating multiply
            v0 *= residz[(long)row * ldc + c0];
            v1 *= residz[(long)row * ldc + c1];
        } else if (residz) {
            v0 += residz[(long)row * ldc + c0];
            v1 += residz[(long)row * ldc + c1];
        }
        C[(long)row * ldc + c0] = v0;
        C[(long)row * ldc + c1] = v1;
    }
}

// ---------------------------------------------------------------------------
// Attention scores (NT GEMM + rel-position gather), per-head via blockIdx.z:
// attn[h][i][j] = 0.125*( q_i.k_j + rq[h][i][ adm[i][j] ] )
// ---------------------------------------------------------------------------
__global__ __launch_bounds__(256) void score_wmma(
    const float* __restrict__ Q, const float* __restrict__ Km,
    const float* __restrict__ rq, const int* __restrict__ adm,
    float* __restrict__ attn) {
    const int h = blockIdx.z;
    Q += h * DKH;
    Km += h * DKH;
    const float* rqh = rq + (long)h * NTOK * NPK;
    attn += (long)h * NTOK * NTOK;
    const int row0 = blockIdx.y * BM;
    const int col0 = blockIdx.x * BN;

    __shared__ alignas(16) _Float16 sA[BM * BK];
    __shared__ alignas(16) _Float16 sB[BN * BK];  // [n][k]

    const int t = threadIdx.x;
    const int lane = t & 31;
    const int wid = t >> 5;
    const int wm = wid & 3;
    const int wn = wid >> 2;

    v8f acc0 = {0.f, 0.f, 0.f, 0.f, 0.f, 0.f, 0.f, 0.f};
    v8f acc1 = acc0;

#pragma unroll
    for (int kt = 0; kt < DKH / BK; ++kt) {  // 2 iterations
        const int k0 = kt * BK;
#pragma unroll
        for (int i = 0; i < 2; ++i) {
            int f = t + i * 256;
            int r = f >> 3, c = (f & 7) << 2;
            float4 d = *(const float4*)(Q + (long)(row0 + r) * DMODEL + k0 + c);
            _Float16* dst = sA + r * BK + c;
            dst[0] = (_Float16)d.x; dst[1] = (_Float16)d.y;
            dst[2] = (_Float16)d.z; dst[3] = (_Float16)d.w;
        }
        // B^T tile: sB[n][k] = Km[(col0+n)][k0+k]  (NT: rows of K are contiguous)
#pragma unroll
        for (int i = 0; i < 2; ++i) {
            int f = t + i * 256;
            int n = f >> 3, c = (f & 7) << 2;
            float4 d = *(const float4*)(Km + (long)(col0 + n) * DMODEL + k0 + c);
            _Float16* dst = sB + n * BK + c;
            dst[0] = (_Float16)d.x; dst[1] = (_Float16)d.y;
            dst[2] = (_Float16)d.z; dst[3] = (_Float16)d.w;
        }
        __syncthreads();

        union { v16h v; uint4 u[2]; } fa, fb0, fb1;
        {
            const int r = wm * 16 + (lane & 15);
            const int kb = (lane < 16) ? 0 : 8;
            const _Float16* p = sA + r * BK;
            fa.u[0] = *(const uint4*)(p + kb);
            fa.u[1] = *(const uint4*)(p + 16 + kb);
        }
        {
            const int kb = (lane < 16) ? 0 : 8;
            const _Float16* p0 = sB + (wn * 32 + (lane & 15)) * BK + kb;
            fb0.u[0] = *(const uint4*)(p0);
            fb0.u[1] = *(const uint4*)(p0 + 16);
            const _Float16* p1 = sB + (wn * 32 + 16 + (lane & 15)) * BK + kb;
            fb1.u[0] = *(const uint4*)(p1);
            fb1.u[1] = *(const uint4*)(p1 + 16);
        }
        acc0 = __builtin_amdgcn_wmma_f32_16x16x32_f16(false, fa.v, false, fb0.v,
                                                      (short)0, acc0, false, false);
        acc1 = __builtin_amdgcn_wmma_f32_16x16x32_f16(false, fa.v, false, fb1.v,
                                                      (short)0, acc1, false, false);
        __syncthreads();
    }

    const int rbase = row0 + wm * 16 + ((lane < 16) ? 0 : 8);
    const int c0 = col0 + wn * 32 + (lane & 15);
    const int c1 = c0 + 16;
#pragma unroll
    for (int r = 0; r < 8; ++r) {
        const int row = rbase + r;
        const long rowN = (long)row * NTOK;
        int p0 = adm[rowN + c0];
        int p1 = adm[rowN + c1];
        float v0 = 0.125f * (acc0[r] + rqh[(long)row * NPK + p0]);
        float v1 = 0.125f * (acc1[r] + rqh[(long)row * NPK + p1]);
        attn[rowN + c0] = v0;
        attn[rowN + c1] = v1;
    }
}

// ---------------------------------------------------------------------------
// In-place softmax over rows of length N=3072; block per (h,i) row.
// Whole row lives in registers (3 float4 per thread): 1 read + 1 write.
// ---------------------------------------------------------------------------
__global__ __launch_bounds__(256) void softmax_kernel(float* __restrict__ attn) {
    float* p = attn + (long)blockIdx.x * NTOK;
    const int t = threadIdx.x;
    __shared__ float red[256];

    float4 d[3];
#pragma unroll
    for (int i = 0; i < 3; ++i) d[i] = *(const float4*)(p + ((t + i * 256) << 2));

    float m = -1e30f;
#pragma unroll
    for (int i = 0; i < 3; ++i) {
        m = fmaxf(m, fmaxf(fmaxf(d[i].x, d[i].y), fmaxf(d[i].z, d[i].w)));
    }
    red[t] = m;
    __syncthreads();
    for (int off = 128; off > 0; off >>= 1) {
        if (t < off) red[t] = fmaxf(red[t], red[t + off]);
        __syncthreads();
    }
    m = red[0];
    __syncthreads();

    float s = 0.0f;
#pragma unroll
    for (int i = 0; i < 3; ++i) {
        d[i].x = __expf(d[i].x - m);
        d[i].y = __expf(d[i].y - m);
        d[i].z = __expf(d[i].z - m);
        d[i].w = __expf(d[i].w - m);
        s += d[i].x + d[i].y + d[i].z + d[i].w;
    }
    red[t] = s;
    __syncthreads();
    for (int off = 128; off > 0; off >>= 1) {
        if (t < off) red[t] += red[t + off];
        __syncthreads();
    }
    const float inv = 1.0f / red[0];
#pragma unroll
    for (int i = 0; i < 3; ++i) {
        d[i].x *= inv; d[i].y *= inv; d[i].z *= inv; d[i].w *= inv;
        *(float4*)(p + ((t + i * 256) << 2)) = d[i];
    }
}

// ---------------------------------------------------------------------------
extern "C" void kernel_launch(void* const* d_in, const int* in_sizes, int n_in,
                              void* d_out, int out_size, void* d_ws, size_t ws_size,
                              hipStream_t stream) {
    (void)in_sizes; (void)n_in; (void)out_size; (void)ws_size;
    const float* x0   = (const float*)d_in[0];
    const int*   adm  = (const int*)d_in[1];
    const float* lns  = (const float*)d_in[2];
    const float* lnb  = (const float*)d_in[3];
    const float* Wq   = (const float*)d_in[4];
    const float* Wk   = (const float*)d_in[5];
    const float* Wv   = (const float*)d_in[6];
    const float* relk = (const float*)d_in[7];
    const float* Wg   = (const float*)d_in[8];
    const float* bg   = (const float*)d_in[9];
    const float* Wo   = (const float*)d_in[10];
    const float* bo   = (const float*)d_in[11];
    const float* lfs  = (const float*)d_in[12];
    const float* lfb  = (const float*)d_in[13];
    const float* W1   = (const float*)d_in[14];
    const float* b1   = (const float*)d_in[15];
    const float* W2   = (const float*)d_in[16];
    const float* b2   = (const float*)d_in[17];

    float* out      = (float*)d_out;
    float* out_x    = out;
    float* out_attn = out + (size_t)NTOK * DMODEL;

    const size_t ND = (size_t)NTOK * DMODEL;
    float* ws = (float*)d_ws;
    float* xn = ws;
    float* q  = ws + 1 * ND;
    float* kk = ws + 2 * ND;
    float* v  = ws + 3 * ND;
    float* o  = ws + 4 * ND;
    float* y  = ws + 5 * ND;
    float* xA = ws + 6 * ND;
    float* xB = ws + 7 * ND;
    float* h1 = ws + 8 * ND;   // 2*ND region (N x 2D)
    float* rq = ws + 10 * ND;  // NH*NTOK*NPK
    float* gate = kk;          // reuse: K dead after score kernel
    float* yn   = xn;          // reuse: xn dead after gate GEMM

    const dim3 blk(256);
    const dim3 g_dd(DMODEL / BN, NTOK / BM, 1);       // 512-col GEMMs
    const dim3 g_f1(2 * DMODEL / BN, NTOK / BM, 1);   // FFN up (1024 cols)
    const dim3 g_sc(NTOK / BN, NTOK / BM, NH);        // scores
    const dim3 g_av(DKH / BN, NTOK / BM, NH);         // attn @ V

    for (int n = 0; n < NL; ++n) {
        const float* xcur = (n == 0) ? x0 : ((n == 1) ? xA : xB);
        float* xnext      = (n == 0) ? xA : ((n == 1) ? xB : out_x);
        float* attn_n = out_attn + (size_t)n * NH * NTOK * NTOK;

        layernorm_kernel<<<NTOK, blk, 0, stream>>>(xcur, lns + n * DMODEL,
                                                   lnb + n * DMODEL, xn);
        // Q, K, V projections
        gemm_wmma<0><<<g_dd, blk, 0, stream>>>(xn, Wq + (size_t)n * DMODEL * DMODEL,
            nullptr, nullptr, q, DMODEL, DMODEL, DMODEL, DMODEL, 0, 0, 0);
        gemm_wmma<0><<<g_dd, blk, 0, stream>>>(xn, Wk + (size_t)n * DMODEL * DMODEL,
            nullptr, nullptr, kk, DMODEL, DMODEL, DMODEL, DMODEL, 0, 0, 0);
        gemm_wmma<0><<<g_dd, blk, 0, stream>>>(xn, Wv + (size_t)n * DMODEL * DMODEL,
            nullptr, nullptr, v, DMODEL, DMODEL, DMODEL, DMODEL, 0, 0, 0);
        // rel-position projection rq[h][i][p]
        relproj_kernel<<<NTOK, 64, 0, stream>>>(q, relk + (size_t)n * NPK * DKH, rq);
        // scores (+gathered rel term, scaled by 1/sqrt(DK)) -> output attn buffer
        score_wmma<<<g_sc, blk, 0, stream>>>(q, kk, rq, adm, attn_n);
        // softmax in place (single read + single write per element)
        softmax_kernel<<<NH * NTOK, blk, 0, stream>>>(attn_n);
        // gate = sigmoid(xn @ Wg + bg)  (reuses K buffer; K dead after scores)
        gemm_wmma<1><<<g_dd, blk, 0, stream>>>(xn, Wg + (size_t)n * DMODEL * DMODEL,
            bg + n * DMODEL, nullptr, gate, DMODEL, DMODEL, DMODEL, DMODEL, 0, 0, 0);
        // o = (attn @ V) * gate   (gate multiply fused in epilogue, per head)
        gemm_wmma<3><<<g_av, blk, 0, stream>>>(attn_n, v, nullptr, gate, o,
            NTOK, NTOK, DMODEL, DMODEL, (long)NTOK * NTOK, DKH, DKH);
        // y = x + (o @ Wo + bo)
        gemm_wmma<0><<<g_dd, blk, 0, stream>>>(o, Wo + (size_t)n * DMODEL * DMODEL,
            bo + n * DMODEL, xcur, y, DMODEL, DMODEL, DMODEL, DMODEL, 0, 0, 0);
        // yn = LN_ff(y)   (reuses xn buffer)
        layernorm_kernel<<<NTOK, blk, 0, stream>>>(y, lfs + n * DMODEL,
                                                   lfb + n * DMODEL, yn);
        // h1 = gelu(yn @ W1 + b1)
        gemm_wmma<2><<<g_f1, blk, 0, stream>>>(yn, W1 + (size_t)n * DMODEL * 2 * DMODEL,
            b1 + (size_t)n * 2 * DMODEL, nullptr, h1,
            DMODEL, DMODEL, 2 * DMODEL, 2 * DMODEL, 0, 0, 0);
        // x_next = x + (h1 @ W2 + b2)   (note: residual from xcur, not y)
        gemm_wmma<0><<<g_dd, blk, 0, stream>>>(h1, W2 + (size_t)n * 2 * DMODEL * DMODEL,
            b2 + n * DMODEL, xcur, xnext,
            2 * DMODEL, 2 * DMODEL, DMODEL, DMODEL, 0, 0, 0);
    }
}
